// HGD_MemNet_65541200937004
// MI455X (gfx1250) — matrix operand order repaired and checked by reading, not compiled, not run on hardware
//
#include <hip/hip_runtime.h>
#include <hip/hip_bf16.h>
#include <stdint.h>

// ---------------- problem constants ----------------
#define B_      32
#define S_T_    64
#define S_REF_  256
#define V_      32000
#define E_      512
#define H_      1024
#define H3_     3072
#define SH_     1024
#define NF_     512
#define NR_     256
#define POOL_   512
#define CONCAT_ 1792

typedef __attribute__((ext_vector_type(16))) __bf16 v16bf;
typedef __attribute__((ext_vector_type(8)))  float  v8f;

union frag_u { v16bf v; uint4 q[2]; };

// ---------------- WMMA fragment helpers (gfx1250 wave32 layouts) ----------------
// A (16x32, MxK), 16-bit: lanes 0-15 row M=lane, K={0..7,16..23}; lanes 16-31 same row, K={8..15,24..31}
__device__ __forceinline__ v16bf load_a_frag(const float* Mp, int ld, int r0, int k0) {
  int lane = threadIdx.x & 31;
  int l = lane & 15, hi = lane >> 4;
  const float* p = Mp + (size_t)(r0 + l) * ld + k0 + hi * 8;
  v16bf f;
#pragma unroll
  for (int i = 0; i < 8; ++i) f[i] = (__bf16)p[i];
#pragma unroll
  for (int i = 0; i < 8; ++i) f[i + 8] = (__bf16)p[i + 16];
  return f;
}

// bf16 A fast path: two aligned 16B loads, no conversion
__device__ __forceinline__ v16bf load_a_frag(const __bf16* Mp, int ld, int r0, int k0) {
  int lane = threadIdx.x & 31;
  int l = lane & 15, hi = lane >> 4;
  const __bf16* p = Mp + (size_t)(r0 + l) * ld + k0 + hi * 8;
  frag_u f;
  f.q[0] = *(const uint4*)p;          // K group 0..7  (8 x bf16 = 16B)
  f.q[1] = *(const uint4*)(p + 16);   // K group 16..23
  return f.v;
}

// B (32x16, KxN) from row-major bf16 weight W (N x K): b[k][n] = W[n][k]
// lanes 0-15: N=lane, K={0..7,16..23} ; lanes 16-31: N=lane-16, K={8..15,24..31}
__device__ __forceinline__ v16bf load_b_frag(const __bf16* W, int ld, int n0, int k0) {
  int lane = threadIdx.x & 31;
  int l = lane & 15, hi = lane >> 4;
  const __bf16* p = W + (size_t)(n0 + l) * ld + k0 + hi * 8;
  frag_u f;
  f.q[0] = *(const uint4*)p;
  f.q[1] = *(const uint4*)(p + 16);
  return f.v;
}

__device__ __forceinline__ v8f wmma_bf16(v16bf a, v16bf b, v8f c) {
  return __builtin_amdgcn_wmma_f32_16x16x32_bf16(false, a, false, b, (short)0, c, false, false);
}

// ---------------- deterministic hash RNG / gumbel ----------------
__device__ __forceinline__ uint32_t hash_u32(uint32_t x) {
  x ^= x >> 16; x *= 0x7feb352dU; x ^= x >> 15; x *= 0x846ca68bU; x ^= x >> 16;
  return x;
}
__device__ __forceinline__ float gumbelf(uint32_t idx, uint32_t seed) {
  uint32_t h = hash_u32(idx * 0x9E3779B9u + seed);
  float u = (float)(h >> 8) * (1.0f / 16777216.0f);   // [0, 1)
  u = fmaxf(u, 1e-12f);
  return -logf(-logf(u));
}
__device__ __forceinline__ float sigmoidf_(float x) { return 1.0f / (1.0f + __expf(-x)); }

// ---------------- weight pre-pack: f32 -> bf16 ----------------
__global__ void cvt_f32_to_bf16(const float* __restrict__ src, __bf16* __restrict__ dst, int n) {
  int i = (blockIdx.x * blockDim.x + threadIdx.x) * 4;
  if (i >= n) return;
  float4 v = *(const float4*)(src + i);
  dst[i + 0] = (__bf16)v.x; dst[i + 1] = (__bf16)v.y;
  dst[i + 2] = (__bf16)v.z; dst[i + 3] = (__bf16)v.w;
}

// ---------------- generic GEMM: C[M,N] = act(A[M,K] @ W[N,K]^T + bias) ----------------
// grid = (M/32, N/64), block = 256 (8 waves); wave -> (mt = w>>2, nt = w&3)
template <typename AT>
__global__ void gemm_bias_act(const AT* __restrict__ A, int lda,
                              const __bf16* __restrict__ W,
                              const float* __restrict__ bias,
                              float* __restrict__ C,
                              int M, int N, int K, int act) {
  int wave = threadIdx.x >> 5;
  int mt = wave >> 2, nt = wave & 3;
  int m0 = blockIdx.x * 32 + mt * 16;
  int n0 = blockIdx.y * 64 + nt * 16;
  v8f acc = {};
  for (int k0 = 0; k0 < K; k0 += 32) {
    if (k0 + 32 < K) __builtin_prefetch(W + (size_t)n0 * K + k0 + 32, 0, 3);
    v16bf a = load_a_frag(A, lda, m0, k0);
    v16bf b = load_b_frag(W, K, n0, k0);
    acc = wmma_bf16(a, b, acc);
  }
  int lane = threadIdx.x & 31;
  int n = n0 + (lane & 15);
  int mb = m0 + ((lane >> 4) << 3);
  float bv = bias ? bias[n] : 0.0f;
#pragma unroll
  for (int r = 0; r < 8; ++r) {
    float v = acc[r] + bv;
    if (act == 1) v = fmaxf(v, 0.0f);
    C[(size_t)(mb + r) * N + n] = v;
  }
}

// ---------------- fused GRU step (one launch per timestep) ----------------
// grid = H/64 = 16 blocks, 256 threads. Each wave accumulates r/z/n strips with 3 WMMAs/K-tile.
// Hidden state dual-format: f32 (for gating hold term) + bf16 (next step's WMMA A operand).
__global__ void gru_step(const float*  __restrict__ h_in_f,   // (B,H) f32
                         const __bf16* __restrict__ h_in_h,   // (B,H) bf16
                         const __bf16* __restrict__ Whh,      // (3H,H) bf16 row-major
                         const float*  __restrict__ bhh,      // (3H)
                         const float*  __restrict__ gi,       // (B,T,3H), bih already added
                         int t, int T,
                         float*  __restrict__ h_out_f,
                         __bf16* __restrict__ h_out_h,
                         float*  __restrict__ ys,             // (B,T,H) f32 or nullptr
                         __bf16* __restrict__ ys_h) {         // (B,T,H) bf16 or nullptr
  int wave = threadIdx.x >> 5;
  int mt = wave >> 2, nt = wave & 3;
  int m0 = mt * 16;                         // batch rows
  int hc0 = blockIdx.x * 64 + nt * 16;      // column within H
  v8f ar = {}, az = {}, an = {};
  for (int k0 = 0; k0 < H_; k0 += 32) {
    if (k0 + 32 < H_) __builtin_prefetch(Whh + (size_t)hc0 * H_ + k0 + 32, 0, 3);
    v16bf a  = load_a_frag(h_in_h, H_, m0, k0);
    v16bf br = load_b_frag(Whh, H_, hc0,          k0);
    v16bf bz = load_b_frag(Whh, H_, H_ + hc0,     k0);
    v16bf bn = load_b_frag(Whh, H_, 2 * H_ + hc0, k0);
    ar = wmma_bf16(a, br, ar);
    az = wmma_bf16(a, bz, az);
    an = wmma_bf16(a, bn, an);
  }
  int lane = threadIdx.x & 31;
  int col = hc0 + (lane & 15);
  int mb = m0 + ((lane >> 4) << 3);
  float br_ = bhh[col], bz_ = bhh[H_ + col], bn_ = bhh[2 * H_ + col];
#pragma unroll
  for (int r = 0; r < 8; ++r) {
    int b = mb + r;
    const float* gih = gi + ((size_t)b * T + t) * H3_;
    float i_r = gih[col], i_z = gih[H_ + col], i_n = gih[2 * H_ + col];
    float rr = sigmoidf_(i_r + ar[r] + br_);
    float zz = sigmoidf_(i_z + az[r] + bz_);
    float nn = tanhf(i_n + rr * (an[r] + bn_));
    float hold = h_in_f[(size_t)b * H_ + col];
    float hnew = (1.0f - zz) * nn + zz * hold;
    h_out_f[(size_t)b * H_ + col] = hnew;
    h_out_h[(size_t)b * H_ + col] = (__bf16)hnew;
    if (ys)   ys  [((size_t)b * T + t) * H_ + col] = hnew;
    if (ys_h) ys_h[((size_t)b * T + t) * H_ + col] = (__bf16)hnew;
  }
}

// ---------------- elementwise / reduction kernels ----------------
__global__ void fill_zero(float* p, int n) {
  int i = blockIdx.x * blockDim.x + threadIdx.x;
  if (i < n) p[i] = 0.0f;
}

__global__ void embed_gather_bf16(const int* __restrict__ ids, const float* __restrict__ table,
                                  __bf16* __restrict__ out) {
  int tok = blockIdx.x;
  int id = ids[tok];
  const float* src = table + (size_t)id * E_;
  __bf16* dst = out + (size_t)tok * E_;
  for (int e = threadIdx.x; e < E_; e += blockDim.x) dst[e] = (__bf16)src[e];
}

__global__ void copy2d(const float* __restrict__ src, int sld,
                       float* __restrict__ dst, int dld, int rows, int cols) {
  int idx = blockIdx.x * blockDim.x + threadIdx.x;
  if (idx >= rows * cols) return;
  int r = idx / cols, c = idx % cols;
  dst[(size_t)r * dld + c] = src[(size_t)r * sld + c];
}

// score[b,s] = Va_b + sum_h Va[h] * tanh(q[b,h] + UaE[b*S+s, h]); one wave per (b,s)
__global__ void attn_scores(const float* __restrict__ q, const float* __restrict__ UaE,
                            const float* __restrict__ Va, const float* __restrict__ Vb,
                            float* __restrict__ scores) {
  int wave = threadIdx.x >> 5, lane = threadIdx.x & 31;
  int idx = blockIdx.x * 8 + wave;          // b*S_REF + s
  int b = idx / S_REF_;
  const float* u = UaE + (size_t)idx * H_;
  const float* qq = q + (size_t)b * H_;
  float acc = 0.0f;
  for (int h = lane; h < H_; h += 32) acc += Va[h] * tanhf(qq[h] + u[h]);
  for (int o = 16; o; o >>= 1) acc += __shfl_xor(acc, o, 32);
  if (lane == 0) scores[idx] = acc + Vb[0];
}

// softmax over cols=256 per row (blockDim must be 256)
__global__ void softmax_rows(float* __restrict__ x) {
  __shared__ float red[256];
  int b = blockIdx.x, tid = threadIdx.x;
  float v = x[b * S_REF_ + tid];
  red[tid] = v; __syncthreads();
  for (int s = 128; s; s >>= 1) { if (tid < s) red[tid] = fmaxf(red[tid], red[tid + s]); __syncthreads(); }
  float mx = red[0]; __syncthreads();
  float e = __expf(v - mx);
  red[tid] = e; __syncthreads();
  for (int s = 128; s; s >>= 1) { if (tid < s) red[tid] += red[tid + s]; __syncthreads(); }
  x[b * S_REF_ + tid] = e / red[0];
}

// ctx[b,h] = sum_s w[b,s] * enc[b,s,h]; grid (B, H/256)
__global__ void attn_context(const float* __restrict__ w, const float* __restrict__ enc,
                             float* __restrict__ ctx) {
  int b = blockIdx.x, h = blockIdx.y * 256 + threadIdx.x;
  float acc = 0.0f;
  for (int s = 0; s < S_REF_; ++s)
    acc += w[b * S_REF_ + s] * enc[((size_t)b * S_REF_ + s) * H_ + h];
  ctx[(size_t)b * H_ + h] = acc;
}

// contrib[b,o] = sum_h softmax_h(Weff[o,h] + gumbel) * hprev[b,h]; one wave per (b,o) row
__global__ void gumbel_contrib(const float* __restrict__ Wf, const float* __restrict__ Wv,
                               const float* __restrict__ hprev, float* __restrict__ contrib) {
  int wave = threadIdx.x >> 5, lane = threadIdx.x & 31;
  int row = blockIdx.x * 8 + wave;          // b*H + o
  int b = row >> 10, o = row & (H_ - 1);
  const float* wrow_f = Wf + (size_t)o * H_;
  const float* wrow_v = Wv + (size_t)o * H_;
  uint32_t base = (uint32_t)row * (uint32_t)H_;
  float mx = -1e30f;
  for (int h = lane; h < H_; h += 32) {
    float l = wrow_f[h] + wrow_v[h] + gumbelf(base + h, 0x243F6A88u);
    mx = fmaxf(mx, l);
  }
  for (int o2 = 16; o2; o2 >>= 1) mx = fmaxf(mx, __shfl_xor(mx, o2, 32));
  float se = 0.0f, num = 0.0f;
  for (int h = lane; h < H_; h += 32) {
    float l = wrow_f[h] + wrow_v[h] + gumbelf(base + h, 0x243F6A88u);
    float e = __expf(l - mx);
    se += e; num += e * hprev[(size_t)b * H_ + h];
  }
  for (int o2 = 16; o2; o2 >>= 1) { se += __shfl_xor(se, o2, 32); num += __shfl_xor(num, o2, 32); }
  if (lane == 0) contrib[row] = num / se;
}

// h_next = tanh(ic + contrib); layernorm -> hnorm (ws) and d_out[0:B*H]
__global__ void tanh_layernorm(const float* __restrict__ ic, const float* __restrict__ ct,
                               const float* __restrict__ g, const float* __restrict__ bb,
                               float* __restrict__ hnorm, float* __restrict__ dout) {
  __shared__ float red[256];
  int b = blockIdx.x, tid = threadIdx.x;
  float v[4]; float s = 0.0f;
#pragma unroll
  for (int i = 0; i < 4; ++i) {
    int h = tid * 4 + i;
    float x = tanhf(ic[(size_t)b * H_ + h] + ct[(size_t)b * H_ + h]);
    v[i] = x; s += x;
  }
  red[tid] = s; __syncthreads();
  for (int st = 128; st; st >>= 1) { if (tid < st) red[tid] += red[tid + st]; __syncthreads(); }
  float mu = red[0] * (1.0f / H_); __syncthreads();
  float s2 = 0.0f;
#pragma unroll
  for (int i = 0; i < 4; ++i) { float d = v[i] - mu; s2 += d * d; }
  red[tid] = s2; __syncthreads();
  for (int st = 128; st; st >>= 1) { if (tid < st) red[tid] += red[tid + st]; __syncthreads(); }
  float inv = rsqrtf(red[0] * (1.0f / H_) + 1e-5f);
#pragma unroll
  for (int i = 0; i < 4; ++i) {
    int h = tid * 4 + i;
    float y = (v[i] - mu) * inv * g[h] + bb[h];
    hnorm[(size_t)b * H_ + h] = y;
    dout[(size_t)b * H_ + h] = y;
  }
}

// key = log_softmax(logits) + gumbel ; rank-by-count -> rnd[b, rank] = pool value.
// grid = B, blockDim = 512. Stable tie-break (lower index first) matches jax top_k.
__global__ void topk_select(const float* __restrict__ logits, const float* __restrict__ hnorm,
                            float* __restrict__ rnd) {
  __shared__ float red[POOL_];
  __shared__ float key[POOL_];
  int b = blockIdx.x, p = threadIdx.x;
  float l = logits[b * POOL_ + p];
  red[p] = l; __syncthreads();
  for (int st = 256; st; st >>= 1) { if (p < st) red[p] = fmaxf(red[p], red[p + st]); __syncthreads(); }
  float mx = red[0]; __syncthreads();
  float e = __expf(l - mx);
  red[p] = e; __syncthreads();
  for (int st = 256; st; st >>= 1) { if (p < st) red[p] += red[p + st]; __syncthreads(); }
  float lse = logf(red[0]) + mx;
  float k = (l - lse) + gumbelf((uint32_t)(b * POOL_ + p), 0xB5297A4Du);
  key[p] = k; __syncthreads();
  int cnt = 0;
  for (int q = 0; q < POOL_; ++q) {
    float kq = key[q];
    cnt += (kq > k) || (kq == k && q < p);
  }
  if (cnt < NR_) rnd[b * NR_ + cnt] = hnorm[(size_t)b * H_ + NF_ + p];
}

// out[b] = sigmoid(sum_k x[b,k]*w[k] + bias); grid = B, blockDim = 256
__global__ void dot_sigmoid(const float* __restrict__ x, const float* __restrict__ w,
                            const float* __restrict__ bias, float* __restrict__ out, int K) {
  __shared__ float red[256];
  int b = blockIdx.x, tid = threadIdx.x;
  float a = 0.0f;
  for (int k = tid; k < K; k += 256) a += x[(size_t)b * K + k] * w[k];
  red[tid] = a; __syncthreads();
  for (int st = 128; st; st >>= 1) { if (tid < st) red[tid] += red[tid + st]; __syncthreads(); }
  if (tid == 0) out[b] = sigmoidf_(red[0] + bias[0]);
}

// ---------------- host-side orchestration ----------------
static inline void cvt_launch(const float* src, __bf16* dst, size_t n, hipStream_t s) {
  cvt_f32_to_bf16<<<(unsigned)((n / 4 + 255) / 256), 256, 0, s>>>(src, dst, (int)n);
}

extern "C" void kernel_launch(void* const* d_in, const int* in_sizes, int n_in,
                              void* d_out, int out_size, void* d_ws, size_t ws_size,
                              hipStream_t stream) {
  (void)in_sizes; (void)n_in; (void)out_size; (void)ws_size;
  const int*   x_t        = (const int*)  d_in[0];
  const int*   x_ref      = (const int*)  d_in[1];
  const float* h_prev     = (const float*)d_in[2];
  const float* embed      = (const float*)d_in[3];
  const float* enc_Wih    = (const float*)d_in[4];
  const float* enc_Whh    = (const float*)d_in[5];
  const float* enc_bih    = (const float*)d_in[6];
  const float* enc_bhh    = (const float*)d_in[7];
  const float* xt_Wih     = (const float*)d_in[8];
  const float* xt_Whh     = (const float*)d_in[9];
  const float* xt_bih     = (const float*)d_in[10];
  const float* xt_bhh     = (const float*)d_in[11];
  const float* Wa_w       = (const float*)d_in[12];
  const float* Wa_b       = (const float*)d_in[13];
  const float* Ua_w       = (const float*)d_in[14];
  const float* Ua_b       = (const float*)d_in[15];
  const float* Va_w       = (const float*)d_in[16];
  const float* Va_b       = (const float*)d_in[17];
  const float* cell_Wih_w = (const float*)d_in[18];
  const float* cell_Wih_b = (const float*)d_in[19];
  const float* W_hh_fixed = (const float*)d_in[20];
  const float* W_hh_virt  = (const float*)d_in[21];
  const float* ln_g       = (const float*)d_in[22];
  const float* ln_b       = (const float*)d_in[23];
  const float* rs_w       = (const float*)d_in[24];
  const float* rs_b       = (const float*)d_in[25];
  const float* gate_w1    = (const float*)d_in[26];
  const float* gate_b1    = (const float*)d_in[27];
  const float* gate_w2    = (const float*)d_in[28];
  const float* gate_b2    = (const float*)d_in[29];
  const float* out_w1     = (const float*)d_in[30];
  const float* out_b1     = (const float*)d_in[31];
  const float* out_w2     = (const float*)d_in[32];
  const float* out_b2     = (const float*)d_in[33];
  float* out = (float*)d_out;

  // workspace bump allocator (float units; every size is a multiple of 4 -> 16B alignment kept)
  float* wsp = (float*)d_ws;
  size_t off = 0;
  auto alloc = [&](size_t nfloats) { float* p = wsp + off; off += nfloats; return p; };
  float*  gi_ref    = alloc((size_t)B_ * S_REF_ * H3_);       // 25.2M f — dead after encoder scan
  float*  gi_t      = alloc((size_t)B_ * S_T_ * H3_);
  float*  enc_out   = alloc((size_t)B_ * S_REF_ * H_);        // f32 (attn ctx)
  __bf16* enc_out_h = (__bf16*)alloc((size_t)B_ * S_REF_ * H_ / 2);  // bf16 (WMMA A for UaE)
  __bf16* emb_ref_h = (__bf16*)alloc((size_t)B_ * S_REF_ * E_ / 2);
  __bf16* emb_t_h   = (__bf16*)alloc((size_t)B_ * S_T_ * E_ / 2);
  // bf16 weight pack
  __bf16* enc_Wih_h = (__bf16*)alloc((size_t)H3_ * E_ / 2);
  __bf16* xt_Wih_h  = (__bf16*)alloc((size_t)H3_ * E_ / 2);
  __bf16* enc_Whh_h = (__bf16*)alloc((size_t)H3_ * H_ / 2);
  __bf16* xt_Whh_h  = (__bf16*)alloc((size_t)H3_ * H_ / 2);
  __bf16* Wa_h      = (__bf16*)alloc((size_t)H_ * H_ / 2);
  __bf16* Ua_h      = (__bf16*)alloc((size_t)H_ * H_ / 2);
  __bf16* cell_h    = (__bf16*)alloc((size_t)H_ * 2 * H_ / 2);
  __bf16* rs_h      = (__bf16*)alloc((size_t)POOL_ * POOL_ / 2);
  __bf16* gate1_h   = (__bf16*)alloc((size_t)SH_ * CONCAT_ / 2);
  __bf16* out1_h    = (__bf16*)alloc((size_t)SH_ * CONCAT_ / 2);
  // hidden-state ping-pong, dual format
  float*  hf0  = alloc((size_t)B_ * H_);
  float*  hf1  = alloc((size_t)B_ * H_);
  __bf16* hh0  = (__bf16*)alloc((size_t)B_ * H_ / 2);
  __bf16* hh1  = (__bf16*)alloc((size_t)B_ * H_ / 2);
  float* q        = alloc((size_t)B_ * H_);
  float* scores   = alloc((size_t)B_ * S_REF_);
  float* ctx      = alloc((size_t)B_ * H_);
  float* rnn_in   = alloc((size_t)B_ * 2 * H_);
  float* ic       = alloc((size_t)B_ * H_);
  float* contrib  = alloc((size_t)B_ * H_);
  float* hnorm    = alloc((size_t)B_ * H_);
  float* rsl      = alloc((size_t)B_ * POOL_);
  float* rnd      = alloc((size_t)B_ * NR_);
  float* combined = alloc((size_t)B_ * CONCAT_);
  float* gateh    = alloc((size_t)B_ * SH_);
  float* outh     = alloc((size_t)B_ * SH_);
  // aliases into gi_ref (dead after encoder scan): UaE (8.39M f) + out_w2 bf16 (16.38M f) <= 25.2M f
  float*  UaE      = gi_ref;
  __bf16* out_w2_h = (__bf16*)(gi_ref + (size_t)B_ * S_REF_ * H_);

  // 0) pre-pack weights needed before/during the scans
  cvt_launch(enc_Wih, enc_Wih_h, (size_t)H3_ * E_, stream);
  cvt_launch(xt_Wih,  xt_Wih_h,  (size_t)H3_ * E_, stream);
  cvt_launch(enc_Whh, enc_Whh_h, (size_t)H3_ * H_, stream);
  cvt_launch(xt_Whh,  xt_Whh_h,  (size_t)H3_ * H_, stream);
  cvt_launch(Wa_w, Wa_h, (size_t)H_ * H_, stream);
  cvt_launch(Ua_w, Ua_h, (size_t)H_ * H_, stream);
  cvt_launch(cell_Wih_w, cell_h, (size_t)H_ * 2 * H_, stream);
  cvt_launch(rs_w, rs_h, (size_t)POOL_ * POOL_, stream);
  cvt_launch(gate_w1, gate1_h, (size_t)SH_ * CONCAT_, stream);
  cvt_launch(out_w1,  out1_h,  (size_t)SH_ * CONCAT_, stream);

  // 1) embeddings (gathered straight to bf16 — only consumed as WMMA A operands)
  embed_gather_bf16<<<B_ * S_REF_, 256, 0, stream>>>(x_ref, embed, emb_ref_h);
  embed_gather_bf16<<<B_ * S_T_,   256, 0, stream>>>(x_t,   embed, emb_t_h);

  // 2) gi = emb @ Wih^T + bih   (bf16 A x bf16 B WMMA)
  gemm_bias_act<__bf16><<<dim3(B_ * S_REF_ / 32, H3_ / 64), 256, 0, stream>>>(
      emb_ref_h, E_, enc_Wih_h, enc_bih, gi_ref, B_ * S_REF_, H3_, E_, 0);
  gemm_bias_act<__bf16><<<dim3(B_ * S_T_ / 32, H3_ / 64), 256, 0, stream>>>(
      emb_t_h, E_, xt_Wih_h, xt_bih, gi_t, B_ * S_T_, H3_, E_, 0);

  // 3) encoder GRU scan (256 fused WMMA steps)
  fill_zero<<<(B_ * H_ + 255) / 256, 256, 0, stream>>>(hf0, B_ * H_);
  fill_zero<<<(B_ * H_ / 2 + 255) / 256, 256, 0, stream>>>((float*)hh0, B_ * H_ / 2);
  for (int t = 0; t < S_REF_; ++t) {
    const float*  hif = (t & 1) ? hf1 : hf0;
    const __bf16* hih = (t & 1) ? hh1 : hh0;
    float*  hof = (t & 1) ? hf0 : hf1;
    __bf16* hoh = (t & 1) ? hh0 : hh1;
    gru_step<<<H_ / 64, 256, 0, stream>>>(hif, hih, enc_Whh_h, enc_bhh, gi_ref, t, S_REF_,
                                          hof, hoh, enc_out, enc_out_h);
  }

  // gi_ref is dead now: pack out_w2 (bf16) into the aliased region
  cvt_launch(out_w2, out_w2_h, (size_t)V_ * SH_, stream);

  // 4) x_t GRU scan (64 steps); final hidden ends in hf0
  fill_zero<<<(B_ * H_ + 255) / 256, 256, 0, stream>>>(hf0, B_ * H_);
  fill_zero<<<(B_ * H_ / 2 + 255) / 256, 256, 0, stream>>>((float*)hh0, B_ * H_ / 2);
  for (int t = 0; t < S_T_; ++t) {
    const float*  hif = (t & 1) ? hf1 : hf0;
    const __bf16* hih = (t & 1) ? hh1 : hh0;
    float*  hof = (t & 1) ? hf0 : hf1;
    __bf16* hoh = (t & 1) ? hh0 : hh1;
    gru_step<<<H_ / 64, 256, 0, stream>>>(hif, hih, xt_Whh_h, xt_bhh, gi_t, t, S_T_,
                                          hof, hoh, nullptr, nullptr);
  }
  float* x_t_enc = hf0;

  // 5) attention
  gemm_bias_act<float><<<dim3(1, H_ / 64), 256, 0, stream>>>(
      h_prev, H_, Wa_h, Wa_b, q, B_, H_, H_, 0);
  gemm_bias_act<__bf16><<<dim3(B_ * S_REF_ / 32, H_ / 64), 256, 0, stream>>>(
      enc_out_h, H_, Ua_h, Ua_b, UaE, B_ * S_REF_, H_, H_, 0);
  attn_scores<<<B_ * S_REF_ / 8, 256, 0, stream>>>(q, UaE, Va_w, Va_b, scores);
  softmax_rows<<<B_, 256, 0, stream>>>(scores);
  attn_context<<<dim3(B_, H_ / 256), 256, 0, stream>>>(scores, enc_out, ctx);

  // 6) rnn_in = [x_t_enc | ctx]; input_contrib
  copy2d<<<(B_ * H_ + 255) / 256, 256, 0, stream>>>(x_t_enc, H_, rnn_in,      2 * H_, B_, H_);
  copy2d<<<(B_ * H_ + 255) / 256, 256, 0, stream>>>(ctx,     H_, rnn_in + H_, 2 * H_, B_, H_);
  gemm_bias_act<float><<<dim3(1, H_ / 64), 256, 0, stream>>>(
      rnn_in, 2 * H_, cell_h, cell_Wih_b, ic, B_, H_, 2 * H_, 0);

  // 7) gumbel-softmax hidden transition + tanh + layernorm
  gumbel_contrib<<<B_ * H_ / 8, 256, 0, stream>>>(W_hh_fixed, W_hh_virt, h_prev, contrib);
  tanh_layernorm<<<B_, 256, 0, stream>>>(ic, contrib, ln_g, ln_b, hnorm, out);

  // 8) random selection over pool
  gemm_bias_act<float><<<dim3(1, POOL_ / 64), 256, 0, stream>>>(
      hnorm + NF_, H_, rs_h, rs_b, rsl, B_, POOL_, POOL_, 0);
  topk_select<<<B_, POOL_, 0, stream>>>(rsl, hnorm, rnd);

  // 9) combined = [fixed | rnd | ctx]
  copy2d<<<(B_ * NF_ + 255) / 256, 256, 0, stream>>>(hnorm, H_,  combined,             CONCAT_, B_, NF_);
  copy2d<<<(B_ * NR_ + 255) / 256, 256, 0, stream>>>(rnd,   NR_, combined + NF_,       CONCAT_, B_, NR_);
  copy2d<<<(B_ * H_  + 255) / 256, 256, 0, stream>>>(ctx,   H_,  combined + NF_ + NR_, CONCAT_, B_, H_);

  // 10) gate head -> out[B*H .. B*H+B)
  gemm_bias_act<float><<<dim3(1, SH_ / 64), 256, 0, stream>>>(
      combined, CONCAT_, gate1_h, gate_b1, gateh, B_, SH_, CONCAT_, 1);
  dot_sigmoid<<<B_, 256, 0, stream>>>(gateh, gate_w2, gate_b2, out + (size_t)B_ * H_, SH_);

  // 11) output head -> out[B*H+B ..)
  gemm_bias_act<float><<<dim3(1, SH_ / 64), 256, 0, stream>>>(
      combined, CONCAT_, out1_h, out_b1, outh, B_, SH_, CONCAT_, 1);
  gemm_bias_act<float><<<dim3(1, V_ / 64), 256, 0, stream>>>(
      outh, SH_, out_w2_h, out_b2, out + (size_t)B_ * H_ + B_, B_, V_, SH_, 0);
}
